// Attention_36077725286527
// MI455X (gfx1250) — compile-verified, matrix operand order checked by reading, not run on hardware
//
#include <hip/hip_runtime.h>
#include <hip/hip_bf16.h>

typedef __attribute__((ext_vector_type(16))) __bf16 v16bf;
typedef __attribute__((ext_vector_type(8)))  float  v8f;

constexpr int  Bn = 8, Sn = 2048, Dn = 768, Fn = 768;
constexpr int  CHUNK = 256;               // query rows per chunk (per batch)
constexpr int  NCHUNK = Sn / CHUNK;       // 8
constexpr float RSQRT_SCALE = 0.03608439182435161f;  // 1/sqrt(768)

// ---------------------------------------------------------------- CDNA5 async copy
// GLOBAL_LOAD_ASYNC_TO_LDS_B128 (GV mode): LDS[vdst + lane] = MEM[vaddr + lane],
// tracked by ASYNCcnt. Low 32 bits of a generic shared pointer are the
// wave-relative LDS address (ISA 10.2 aperture mapping), so truncation is the
// correct VDST value.
__device__ __forceinline__ void async_b128(const __bf16* gsrc, const __bf16* ldst) {
  unsigned d = (unsigned)(unsigned long long)ldst;
  asm volatile("global_load_async_to_lds_b128 %0, %1, off"
               :: "v"(d), "v"(gsrc) : "memory");
}
__device__ __forceinline__ void wait_async0() {
  asm volatile("s_wait_asynccnt 0x0" ::: "memory");
}

// ---------------------------------------------------------------- convert
__global__ __launch_bounds__(256) void cvt_bf16(const float* __restrict__ in,
                                                __bf16* __restrict__ out, int n) {
  int i = blockIdx.x * blockDim.x + threadIdx.x;
  int stride = gridDim.x * blockDim.x;
  for (; i < n; i += stride) out[i] = (__bf16)in[i];
}

// ---------------------------------------------------------------- row norms
__global__ __launch_bounds__(256) void row_norm2(const __bf16* __restrict__ a,
                                                 float* __restrict__ out,
                                                 int rows, int cols) {
  int wave = threadIdx.x >> 5, lane = threadIdx.x & 31;
  int row = blockIdx.x * 8 + wave;
  if (row >= rows) return;
  const __bf16* p = a + (long)row * cols;
  float s = 0.f;
  for (int f = lane; f < cols; f += 32) { float v = (float)p[f]; s += v * v; }
  #pragma unroll
  for (int off = 16; off > 0; off >>= 1) s += __shfl_down(s, off, 32);
  if (lane == 0) out[row] = s;
}

// ---------------------------------------------------------------- WMMA tile helpers
// Block tile 128x128, BK=32, 256 threads = 8 waves, each wave does 32x64.
// Double-buffered LDS stages; async global->LDS copies for stage k+1 are issued
// before the WMMAs on stage k; one s_wait_asynccnt + barrier per iteration.
// A frag (16x32 bf16): lane holds row (lane&15), K half (lane>>4)*16..+16.
// B frag (LDS stored [n][k]): lane holds col (lane&15), K half (lane>>4)*16.
// C layout: col = lane&15, row = (lane>>4)*8 + vgpr.

#define GEMM_PROLOG()                                                   \
  __shared__ __align__(32) __bf16 lA[2][128 * 32];                      \
  __shared__ __align__(32) __bf16 lB[2][128 * 32];                      \
  const int tid  = threadIdx.x;                                         \
  const int lane = tid & 31;                                            \
  const int w    = tid >> 5;                                            \
  const int wm   = w & 3;   /* 4 waves along M: 4*32 = 128 */           \
  const int wn   = w >> 2;  /* 2 waves along N: 2*64 = 128 */           \
  v8f acc[2][4];                                                        \
  {                                                                     \
    v8f z = {0.f,0.f,0.f,0.f,0.f,0.f,0.f,0.f};                          \
    _Pragma("unroll") for (int i = 0; i < 2; ++i)                       \
      _Pragma("unroll") for (int j = 0; j < 4; ++j) acc[i][j] = z;      \
  }                                                                     \
  const int rowL = tid >> 1, half = tid & 1;                            \
  const int sOff = rowL * 32 + half * 16;

#define MMA_STEP(st)                                                    \
  do {                                                                  \
    v16bf afr[2], bfr[4];                                               \
    _Pragma("unroll")                                                   \
    for (int i = 0; i < 2; ++i)                                         \
      afr[i] = *(const v16bf*)&lA[st][(wm * 32 + i * 16 + (lane & 15))  \
                                     * 32 + (lane >> 4) * 16];          \
    _Pragma("unroll")                                                   \
    for (int j = 0; j < 4; ++j)                                         \
      bfr[j] = *(const v16bf*)&lB[st][(wn * 64 + j * 16 + (lane & 15))  \
                                      * 32 + (lane >> 4) * 16];         \
    _Pragma("unroll")                                                   \
    for (int i = 0; i < 2; ++i)                                         \
      _Pragma("unroll")                                                 \
      for (int j = 0; j < 4; ++j)                                       \
        acc[i][j] = __builtin_amdgcn_wmma_f32_16x16x32_bf16(            \
            false, afr[i], false, bfr[j], (short)0, acc[i][j],          \
            false, false);                                              \
  } while (0)

// ---------------------------------------------------------------- QKV projection
// C[m,n] = sum_d A[m,d] * W[n,d]   (NT: both K-contiguous), bf16 output
__global__ __launch_bounds__(256) void gemm_nt_bf16out(
    const __bf16* __restrict__ A, const __bf16* __restrict__ Bm,
    __bf16* __restrict__ C, int K, int ldc) {
  GEMM_PROLOG();
  const int m0 = blockIdx.y * 128, n0 = blockIdx.x * 128;
  const __bf16* pA = A + (long)(m0 + rowL) * K + half * 16;
  const __bf16* pB = Bm + (long)(n0 + rowL) * K + half * 16;
  async_b128(pA,     &lA[0][sOff]);
  async_b128(pA + 8, &lA[0][sOff + 8]);
  async_b128(pB,     &lB[0][sOff]);
  async_b128(pB + 8, &lB[0][sOff + 8]);
  wait_async0();
  __syncthreads();
  int cur = 0;
  for (int k0 = 32; k0 < K; k0 += 32) {
    const int nxt = cur ^ 1;
    async_b128(pA + k0,     &lA[nxt][sOff]);
    async_b128(pA + k0 + 8, &lA[nxt][sOff + 8]);
    async_b128(pB + k0,     &lB[nxt][sOff]);
    async_b128(pB + k0 + 8, &lB[nxt][sOff + 8]);
    __builtin_prefetch(pA + k0 + 32, 0, 3);
    __builtin_prefetch(pB + k0 + 32, 0, 3);
    MMA_STEP(cur);
    wait_async0();
    __syncthreads();
    cur = nxt;
  }
  MMA_STEP(cur);
  #pragma unroll
  for (int i = 0; i < 2; ++i)
    #pragma unroll
    for (int j = 0; j < 4; ++j)
      #pragma unroll
      for (int r = 0; r < 8; ++r) {
        int row = m0 + wm * 32 + i * 16 + (lane >> 4) * 8 + r;
        int col = n0 + wn * 64 + j * 16 + (lane & 15);
        C[(long)row * ldc + col] = (__bf16)acc[i][j][r];
      }
}

// ---------------------------------------------------------------- distance scores
// scores[m,n] = sqrt(max(q2[m]+k2[n]-2*q.k, 0)) / sqrt(768), fp32, per chunk
__global__ __launch_bounds__(256) void dist_scores(
    const __bf16* __restrict__ qh, const __bf16* __restrict__ kh,
    const float* __restrict__ q2, const float* __restrict__ k2,
    float* __restrict__ scores, int c0) {
  GEMM_PROLOG();
  const int b = blockIdx.z;
  const int m0 = blockIdx.y * 128, n0 = blockIdx.x * 128;
  const __bf16* pA = qh + ((long)b * Sn + c0 + m0 + rowL) * Fn + half * 16;
  const __bf16* pB = kh + ((long)b * Sn + n0 + rowL) * Fn + half * 16;
  float* Sc = scores + (long)b * CHUNK * Sn;
  async_b128(pA,     &lA[0][sOff]);
  async_b128(pA + 8, &lA[0][sOff + 8]);
  async_b128(pB,     &lB[0][sOff]);
  async_b128(pB + 8, &lB[0][sOff + 8]);
  wait_async0();
  __syncthreads();
  int cur = 0;
  for (int k0 = 32; k0 < Fn; k0 += 32) {
    const int nxt = cur ^ 1;
    async_b128(pA + k0,     &lA[nxt][sOff]);
    async_b128(pA + k0 + 8, &lA[nxt][sOff + 8]);
    async_b128(pB + k0,     &lB[nxt][sOff]);
    async_b128(pB + k0 + 8, &lB[nxt][sOff + 8]);
    __builtin_prefetch(pA + k0 + 32, 0, 3);
    __builtin_prefetch(pB + k0 + 32, 0, 3);
    MMA_STEP(cur);
    wait_async0();
    __syncthreads();
    cur = nxt;
  }
  MMA_STEP(cur);
  #pragma unroll
  for (int i = 0; i < 2; ++i)
    #pragma unroll
    for (int j = 0; j < 4; ++j)
      #pragma unroll
      for (int r = 0; r < 8; ++r) {
        int row = m0 + wm * 32 + i * 16 + (lane >> 4) * 8 + r;  // 0..CHUNK-1
        int col = n0 + wn * 64 + j * 16 + (lane & 15);          // 0..Sn-1
        float qk = acc[i][j][r];
        float d2 = q2[(long)b * Sn + c0 + row] + k2[(long)b * Sn + col] - 2.f * qk;
        Sc[(long)row * Sn + col] = sqrtf(fmaxf(d2, 0.f)) * RSQRT_SCALE;
      }
}

// ---------------------------------------------------------------- softmax rows
// one block per row of length Sn; writes bf16 probabilities
__global__ __launch_bounds__(256) void softmax_rows(
    const float* __restrict__ scores, __bf16* __restrict__ probs) {
  const long row = blockIdx.x;
  const float* src = scores + row * Sn;
  __bf16* dst = probs + row * Sn;
  __shared__ float red[8];
  const int tid = threadIdx.x, lane = tid & 31, w = tid >> 5;
  float vals[8];
  float m = -1e30f;
  #pragma unroll
  for (int i = 0; i < 8; ++i) { vals[i] = src[tid + i * 256]; m = fmaxf(m, vals[i]); }
  #pragma unroll
  for (int off = 16; off > 0; off >>= 1) m = fmaxf(m, __shfl_down(m, off, 32));
  if (lane == 0) red[w] = m;
  __syncthreads();
  float mAll = red[0];
  #pragma unroll
  for (int i = 1; i < 8; ++i) mAll = fmaxf(mAll, red[i]);
  __syncthreads();
  float s = 0.f;
  #pragma unroll
  for (int i = 0; i < 8; ++i) { vals[i] = __expf(vals[i] - mAll); s += vals[i]; }
  #pragma unroll
  for (int off = 16; off > 0; off >>= 1) s += __shfl_down(s, off, 32);
  if (lane == 0) red[w] = s;
  __syncthreads();
  float sAll = 0.f;
  #pragma unroll
  for (int i = 0; i < 8; ++i) sAll += red[i];
  float inv = 1.f / sAll;
  #pragma unroll
  for (int i = 0; i < 8; ++i) dst[tid + i * 256] = (__bf16)(vals[i] * inv);
}

// ---------------------------------------------------------------- attn @ V
// out[m,f] = sum_j P[m,j] V[j,f];  A NT async, B NN ([K][N]) transposed via LDS
__global__ __launch_bounds__(256) void attn_v(
    const __bf16* __restrict__ probs, const __bf16* __restrict__ vh,
    float* __restrict__ out, int c0) {
  GEMM_PROLOG();
  const int b = blockIdx.z;
  const int m0 = blockIdx.y * 128, n0 = blockIdx.x * 128;
  const int kr = tid >> 3, nc = tid & 7;              // B-tile: 32 k-rows x 8 n-chunks
  const __bf16* pA = probs + ((long)b * CHUNK + m0 + rowL) * Sn + half * 16;
  const __bf16* pB = vh + ((long)b * Sn + kr) * Fn + n0 + nc * 16;
  float* C = out + ((long)b * Sn + c0) * Fn;

  async_b128(pA,     &lA[0][sOff]);
  async_b128(pA + 8, &lA[0][sOff + 8]);
  {
    uint4 b0 = ((const uint4*)pB)[0], b1 = ((const uint4*)pB)[1];
    __bf16 tmp[16];
    *(uint4*)&tmp[0] = b0; *(uint4*)&tmp[8] = b1;
    #pragma unroll
    for (int i = 0; i < 16; ++i) lB[0][(nc * 16 + i) * 32 + kr] = tmp[i];
  }
  wait_async0();
  __syncthreads();
  int cur = 0;
  for (int k0 = 32; k0 < Sn; k0 += 32) {
    const int nxt = cur ^ 1;
    async_b128(pA + k0,     &lA[nxt][sOff]);
    async_b128(pA + k0 + 8, &lA[nxt][sOff + 8]);
    const uint4* qb = (const uint4*)(pB + (long)k0 * Fn);
    uint4 b0 = qb[0], b1 = qb[1];
    __builtin_prefetch(pA + k0 + 32, 0, 3);
    __builtin_prefetch(pB + (long)(k0 + 32) * Fn, 0, 3);
    MMA_STEP(cur);
    {
      __bf16 tmp[16];
      *(uint4*)&tmp[0] = b0; *(uint4*)&tmp[8] = b1;
      #pragma unroll
      for (int i = 0; i < 16; ++i) lB[nxt][(nc * 16 + i) * 32 + kr] = tmp[i];
    }
    wait_async0();
    __syncthreads();
    cur = nxt;
  }
  MMA_STEP(cur);
  #pragma unroll
  for (int i = 0; i < 2; ++i)
    #pragma unroll
    for (int j = 0; j < 4; ++j)
      #pragma unroll
      for (int r = 0; r < 8; ++r) {
        int row = m0 + wm * 32 + i * 16 + (lane >> 4) * 8 + r;
        int col = n0 + wn * 64 + j * 16 + (lane & 15);
        C[(long)row * Fn + col] = acc[i][j][r];
      }
}

// ---------------------------------------------------------------- launch
extern "C" void kernel_launch(void* const* d_in, const int* in_sizes, int n_in,
                              void* d_out, int out_size, void* d_ws, size_t ws_size,
                              hipStream_t stream) {
  const float* x  = (const float*)d_in[0];
  const float* Wq = (const float*)d_in[1];
  const float* Wk = (const float*)d_in[2];
  const float* Wv = (const float*)d_in[3];
  (void)in_sizes; (void)n_in; (void)out_size; (void)ws_size;

  const long MS = (long)Bn * Sn;          // 16384 rows
  char* ws = (char*)d_ws;
  __bf16* xh  = (__bf16*)ws;                     // MS*Dn
  __bf16* wqh = xh  + MS * Dn;
  __bf16* wkh = wqh + (long)Fn * Dn;
  __bf16* wvh = wkh + (long)Fn * Dn;
  __bf16* qh  = wvh + (long)Fn * Dn;             // MS*Fn each
  __bf16* kh  = qh + MS * Fn;
  __bf16* vh  = kh + MS * Fn;
  float*  q2  = (float*)(vh + MS * Fn);
  float*  k2  = q2 + MS;
  float*  scores = k2 + MS;                      // Bn*CHUNK*Sn fp32
  __bf16* probs  = (__bf16*)(scores + (long)Bn * CHUNK * Sn);
  float*  out = (float*)d_out;

  // 1) fp32 -> bf16
  cvt_bf16<<<2048, 256, 0, stream>>>(x,  xh,  (int)(MS * Dn));
  cvt_bf16<<<512,  256, 0, stream>>>(Wq, wqh, Fn * Dn);
  cvt_bf16<<<512,  256, 0, stream>>>(Wk, wkh, Fn * Dn);
  cvt_bf16<<<512,  256, 0, stream>>>(Wv, wvh, Fn * Dn);

  // 2) QKV projections: [16384 x 768] @ [768 x 768]^T
  dim3 gP(Fn / 128, (unsigned)(MS / 128));       // (6, 128)
  gemm_nt_bf16out<<<gP, 256, 0, stream>>>(xh, wqh, qh, Dn, Fn);
  gemm_nt_bf16out<<<gP, 256, 0, stream>>>(xh, wkh, kh, Dn, Fn);
  gemm_nt_bf16out<<<gP, 256, 0, stream>>>(xh, wvh, vh, Dn, Fn);

  // 3) row norms
  row_norm2<<<(unsigned)(MS / 8), 256, 0, stream>>>(qh, q2, (int)MS, Fn);
  row_norm2<<<(unsigned)(MS / 8), 256, 0, stream>>>(kh, k2, (int)MS, Fn);

  // 4) chunked distance-attention (chunks serialize on the stream, score buf reused)
  for (int c = 0; c < NCHUNK; ++c) {
    dim3 gS(Sn / 128, CHUNK / 128, Bn);          // (16, 2, 8)
    dist_scores<<<gS, 256, 0, stream>>>(qh, kh, q2, k2, scores, c * CHUNK);
    softmax_rows<<<Bn * CHUNK, 256, 0, stream>>>(scores, probs);
    dim3 gA(Fn / 128, CHUNK / 128, Bn);          // (6, 2, 8)
    attn_v<<<gA, 256, 0, stream>>>(probs, vh, out, c * CHUNK);
  }
}
